// GraphAE_69277822484550
// MI455X (gfx1250) — matrix operand order repaired and checked by reading, not compiled, not run on hardware
//
#include <hip/hip_runtime.h>
#include <hip/hip_bf16.h>
#include <math.h>

// ---------------------------------------------------------------------------
// GraphAE on MI455X (gfx1250, wave32):
//   graph branch : atomics-based mean aggregation + small VALU GEMMs
//   encoder/decoder GEMMs : v_wmma_f32_16x16x32_f16 (f16 in, f32 accum)
//   encoder: double-buffered LDS staging, one barrier per K-step,
//            dual accumulators (independent WMMA chains), branch-free hot loop
// ---------------------------------------------------------------------------

typedef __attribute__((ext_vector_type(16))) _Float16 v16h;
typedef __attribute__((ext_vector_type(8)))  _Float16 v8h;
typedef __attribute__((ext_vector_type(8)))  float    v8f;

#define N_NODES 50000
#define M_ITEMS 5000
#define D_FEAT  64
#define D_HID   32
#define D_EMB   64
#define N_EDGES 1200000
#define BATCH   4096
#define KP      5024   // M_ITEMS padded to multiple of 32 (encoder K), 157 steps
#define NP      5008   // M_ITEMS padded to multiple of 16 (decoder N)

#define WMMA_F16(A, B, C) \
  __builtin_amdgcn_wmma_f32_16x16x32_f16(false, (A), false, (B), (short)0, (C), false, false)

// ---------------------------------------------------------------- utilities
__global__ void k_zero(float* __restrict__ p, int n) {
  int t = blockIdx.x * 256 + threadIdx.x;
  if (t < n) p[t] = 0.0f;
}

__global__ void k_cvt_wenc(const float* __restrict__ W, _Float16* __restrict__ Wh) {
  int t = blockIdx.x * 256 + threadIdx.x;      // t < 64*KP (exact grid)
  int o = t / KP, k = t % KP;
  Wh[t] = (k < M_ITEMS) ? (_Float16)W[o * M_ITEMS + k] : (_Float16)0.0f;
}

__global__ void k_cvt_wdec(const float* __restrict__ W, _Float16* __restrict__ Wh) {
  int t = blockIdx.x * 256 + threadIdx.x;      // t < NP*64 (exact grid)
  int n = t >> 6, k = t & 63;
  Wh[t] = (n < M_ITEMS) ? (_Float16)W[n * 64 + k] : (_Float16)0.0f;
}

// ------------------------------------------------------------ graph branch
__global__ void k_scatter1(const int* __restrict__ src, const int* __restrict__ dst,
                           const float* __restrict__ nx, float* __restrict__ agg1,
                           float* __restrict__ deg) {
  long long t = (long long)blockIdx.x * 256 + threadIdx.x;
  int e = (int)(t >> 6), f = (int)(t & 63);
  int s = src[e], d = dst[e];
  atomicAdd(&agg1[d * 64 + f], nx[s * 64 + f]);
  if (f == 0) atomicAdd(&deg[d], 1.0f);
}

__global__ void k_layer1(const float* __restrict__ agg1, const float* __restrict__ nx,
                         const float* __restrict__ deg, const float* __restrict__ Wl1,
                         const float* __restrict__ bl1, const float* __restrict__ Wr1,
                         float* __restrict__ h) {
  int t = blockIdx.x * 256 + threadIdx.x;      // exact: N_NODES*32
  int o = t & 31, i = t >> 5;
  float rdeg = 1.0f / fmaxf(deg[i], 1.0f);
  const float* ar = agg1 + i * 64;
  const float* xr = nx + i * 64;
  const float* wl = Wl1 + o * 64;
  const float* wr = Wr1 + o * 64;
  float sa = 0.0f, sx = 0.0f;
#pragma unroll 8
  for (int k = 0; k < 64; ++k) { sa += ar[k] * wl[k]; sx += xr[k] * wr[k]; }
  h[i * 32 + o] = fmaxf(bl1[o] + rdeg * sa + sx, 0.0f);
}

__global__ void k_scatter2(const int* __restrict__ src, const int* __restrict__ dst,
                           const float* __restrict__ h, float* __restrict__ agg2) {
  long long t = (long long)blockIdx.x * 256 + threadIdx.x;
  int e = (int)(t >> 5), f = (int)(t & 31);
  atomicAdd(&agg2[dst[e] * 32 + f], h[src[e] * 32 + f]);
}

__global__ void k_layer2(const int* __restrict__ x, const float* __restrict__ agg2,
                         const float* __restrict__ h, const float* __restrict__ deg,
                         const float* __restrict__ Wl2, const float* __restrict__ bl2,
                         const float* __restrict__ Wr2, float* __restrict__ eg) {
  int t = blockIdx.x * 256 + threadIdx.x;      // exact: BATCH*64
  int o = t & 63, b = t >> 6;
  int i = x[b];
  float rdeg = 1.0f / fmaxf(deg[i], 1.0f);
  const float* ar = agg2 + i * 32;
  const float* hr = h + i * 32;
  const float* wl = Wl2 + o * 32;
  const float* wr = Wr2 + o * 32;
  float sa = 0.0f, sh = 0.0f;
#pragma unroll
  for (int k = 0; k < 32; ++k) { sa += ar[k] * wl[k]; sh += hr[k] * wr[k]; }
  eg[b * 64 + o] = bl2[o] + rdeg * sa + sh;
}

// ------------------------------------------------------- WMMA encoder GEMM
// z[4096,64] = sigmoid(emb_graph + items @ W_enc^T + b_enc)
// block = 128 threads (4 waves) = one 16-row M-tile; wave w owns N-tile 16w.
// A gathered+converted into double-buffered LDS; 157 K-steps (154 in a
// branch-free pair loop on acc0/acc1, 3 peeled incl. the guarded tail).
__global__ void __launch_bounds__(128)
k_encoder(const int* __restrict__ x, const float* __restrict__ rating,
          const _Float16* __restrict__ Wh, const float* __restrict__ b_enc,
          const float* __restrict__ eg, float* __restrict__ z) {
  __shared__ int sRow[16];
  __shared__ __align__(16) _Float16 sA[2][16 * 32];
  const int tid  = threadIdx.x;
  const int m0   = blockIdx.x * 16;
  const int lane = tid & 31, wave = tid >> 5;
  const int n0   = wave * 16;
  if (tid < 16) sRow[tid] = x[m0 + tid];
  __syncthreads();

  const int m  = lane & 15;     // A row / B col within tile
  const int h2 = lane >> 4;     // lane half-group
  const int fr = tid >> 3;      // A-fill row (0..15)
  const int fc = (tid & 7) * 4; // A-fill col base (0..28)

  // loop-invariant: gather base for this thread's fill row (hoisted from loop)
  const float* rp = rating + (size_t)sRow[fr] * M_ITEMS + fc;
  _Float16* sp0 = &sA[0][fr * 32 + fc];
  _Float16* sp1 = &sA[1][fr * 32 + fc];
  const _Float16* a0lo = &sA[0][m * 32 + 8 * h2];
  const _Float16* a1lo = &sA[1][m * 32 + 8 * h2];
  const _Float16* bbase = Wh + (size_t)(n0 + m) * KP + 8 * h2;

#define FILL_FAST(KT, SP)                                            \
  do {                                                               \
    float4 v_ = *(const float4*)(rp + (KT) * 32);                    \
    (SP)[0] = (_Float16)v_.x; (SP)[1] = (_Float16)v_.y;              \
    (SP)[2] = (_Float16)v_.z; (SP)[3] = (_Float16)v_.w;              \
  } while (0)

#define LOAD_FRAGS(AF, BF, ALO, S)                                   \
  do {                                                               \
    (AF).h[0] = *(const v8h*)(ALO);                                  \
    (AF).h[1] = *(const v8h*)((ALO) + 16);                           \
    const _Float16* bp_ = bbase + (S) * 32;                          \
    (BF).h[0] = *(const v8h*)(bp_);                                  \
    (BF).h[1] = *(const v8h*)(bp_ + 16);                             \
  } while (0)

  union frag { v16h v; v8h h[2]; };
  v8f acc0 = {}, acc1 = {};

  FILL_FAST(0, sp0);
  __syncthreads();

  // pairs covering steps 0..153 (all fills fully in-bounds: kg < 4992)
  for (int s = 0; s < 154; s += 2) {
    frag af, bf;
    LOAD_FRAGS(af, bf, a0lo, s);
    FILL_FAST(s + 1, sp1);
    __syncthreads();
    acc0 = WMMA_F16(af.v, bf.v, acc0);

    frag ag, bg;
    LOAD_FRAGS(ag, bg, a1lo, s + 1);
    FILL_FAST(s + 2, sp0);
    __syncthreads();
    acc1 = WMMA_F16(ag.v, bg.v, acc1);
  }
  {  // step 154 (buf0), fill 155 -> buf1
    frag af, bf;
    LOAD_FRAGS(af, bf, a0lo, 154);
    FILL_FAST(155, sp1);
    __syncthreads();
    acc0 = WMMA_F16(af.v, bf.v, acc0);
  }
  {  // step 155 (buf1), guarded tail fill 156 -> buf0 (K 4992..5023)
    frag af, bf;
    LOAD_FRAGS(af, bf, a1lo, 155);
    int kg = 156 * 32 + fc;
    float vx = (kg + 0 < M_ITEMS) ? rp[156 * 32 + 0] : 0.0f;
    float vy = (kg + 1 < M_ITEMS) ? rp[156 * 32 + 1] : 0.0f;
    float vz = (kg + 2 < M_ITEMS) ? rp[156 * 32 + 2] : 0.0f;
    float vw = (kg + 3 < M_ITEMS) ? rp[156 * 32 + 3] : 0.0f;
    sp0[0] = (_Float16)vx; sp0[1] = (_Float16)vy;
    sp0[2] = (_Float16)vz; sp0[3] = (_Float16)vw;
    __syncthreads();
    acc1 = WMMA_F16(af.v, bf.v, acc1);
  }
  {  // step 156 (buf0), no fill
    frag af, bf;
    LOAD_FRAGS(af, bf, a0lo, 156);
    acc0 = WMMA_F16(af.v, bf.v, acc0);
  }
  v8f acc = acc0 + acc1;

  const int o = n0 + m;                        // output feature (0..63)
#pragma unroll
  for (int j = 0; j < 8; ++j) {
    int br = m0 + j + 8 * h2;                  // batch row
    float val = eg[br * 64 + o] + acc[j] + b_enc[o];
    z[br * 64 + o] = 1.0f / (1.0f + expf(-val));
  }
#undef FILL_FAST
#undef LOAD_FRAGS
}

// ------------------------------------------------------- WMMA decoder GEMM
// out[4096,5000] = sigmoid(z @ W_dec^T + b_dec); K=64 = 2 independent WMMAs.
__global__ void __launch_bounds__(256)
k_decoder(const float* __restrict__ z, const _Float16* __restrict__ Wh,
          const float* __restrict__ b_dec, float* __restrict__ out) {
  const int tid  = threadIdx.x;
  const int lane = tid & 31, wave = tid >> 5;
  const int tile = blockIdx.x * 8 + wave;      // exact: 256*313 tiles
  const int mt = tile / (NP / 16), nt = tile % (NP / 16);
  const int m0 = mt * 16, n0 = nt * 16;
  const int m  = lane & 15;
  const int h2 = lane >> 4;

  union frag { v16h v; v8h h[2]; };
  union fragf { v16h v; _Float16 e[16]; };
  v8f zero = {};
  v8f part[2];

#pragma unroll
  for (int kt = 0; kt < 2; ++kt) {
    // A fragment from f32 z: row m0+m, K = kt*32 + 8*h2 + [0..7] and +16
    const float* zp = z + (m0 + m) * 64 + kt * 32 + 8 * h2;
    float4 p0 = *(const float4*)(zp);
    float4 p1 = *(const float4*)(zp + 4);
    float4 p2 = *(const float4*)(zp + 16);
    float4 p3 = *(const float4*)(zp + 20);
    fragf af;
    af.e[0]  = (_Float16)p0.x; af.e[1]  = (_Float16)p0.y;
    af.e[2]  = (_Float16)p0.z; af.e[3]  = (_Float16)p0.w;
    af.e[4]  = (_Float16)p1.x; af.e[5]  = (_Float16)p1.y;
    af.e[6]  = (_Float16)p1.z; af.e[7]  = (_Float16)p1.w;
    af.e[8]  = (_Float16)p2.x; af.e[9]  = (_Float16)p2.y;
    af.e[10] = (_Float16)p2.z; af.e[11] = (_Float16)p2.w;
    af.e[12] = (_Float16)p3.x; af.e[13] = (_Float16)p3.y;
    af.e[14] = (_Float16)p3.z; af.e[15] = (_Float16)p3.w;
    // B fragment: col n0+m of padded W_dec (row-major [NP][64])
    frag bf;
    const _Float16* bp = Wh + (size_t)(n0 + m) * 64 + kt * 32 + 8 * h2;
    bf.h[0] = *(const v8h*)(bp);
    bf.h[1] = *(const v8h*)(bp + 16);
    part[kt] = WMMA_F16(af.v, bf.v, zero);
  }
  v8f acc = part[0] + part[1];

  const int o = n0 + m;
  if (o < M_ITEMS) {
    float bd = b_dec[o];
#pragma unroll
    for (int j = 0; j < 8; ++j) {
      int br = m0 + j + 8 * h2;
      float val = acc[j] + bd;
      out[(size_t)br * M_ITEMS + o] = 1.0f / (1.0f + expf(-val));
    }
  }
}

// ------------------------------------------------------------------ launch
extern "C" void kernel_launch(void* const* d_in, const int* in_sizes, int n_in,
                              void* d_out, int out_size, void* d_ws, size_t ws_size,
                              hipStream_t stream) {
  const int*   x      = (const int*)  d_in[0];
  const float* node_x = (const float*)d_in[1];
  const int*   eidx   = (const int*)  d_in[2];
  const float* rating = (const float*)d_in[3];
  const float* W_enc  = (const float*)d_in[4];
  const float* b_enc  = (const float*)d_in[5];
  const float* W_dec  = (const float*)d_in[6];
  const float* b_dec  = (const float*)d_in[7];
  const float* Wl1    = (const float*)d_in[8];
  const float* bl1    = (const float*)d_in[9];
  const float* Wr1    = (const float*)d_in[10];
  const float* Wl2    = (const float*)d_in[11];
  const float* bl2    = (const float*)d_in[12];
  const float* Wr2    = (const float*)d_in[13];
  const int* src = eidx;
  const int* dst = eidx + N_EDGES;
  float* out = (float*)d_out;

  // workspace layout (all region sizes multiples of 64 floats => 256B align)
  float* w = (float*)d_ws;
  size_t off = 0;
  float* deg  = w + off; off += 50048;                 // N_NODES padded
  float* agg1 = w + off; off += (size_t)N_NODES * 64;
  float* agg2 = w + off; off += (size_t)N_NODES * 32;
  float* h    = w + off; off += (size_t)N_NODES * 32;
  float* eg   = w + off; off += (size_t)BATCH * 64;
  float* z    = w + off; off += (size_t)BATCH * 64;
  _Float16* WencH = (_Float16*)(w + off); off += (size_t)(64 * KP) / 2;
  _Float16* WdecH = (_Float16*)(w + off); off += (size_t)(NP * 64) / 2;
  // total ~29.2 MB of the provided workspace

  // zero deg | agg1 | agg2 (contiguous)
  const int zc = 50048 + N_NODES * 64 + N_NODES * 32;
  k_zero<<<(zc + 255) / 256, 256, 0, stream>>>(w, zc);

  // pre-convert weights to padded f16
  k_cvt_wenc<<<(64 * KP) / 256, 256, 0, stream>>>(W_enc, WencH);
  k_cvt_wdec<<<(NP * 64) / 256, 256, 0, stream>>>(W_dec, WdecH);

  // graph branch
  k_scatter1<<<(N_EDGES * 64) / 256, 256, 0, stream>>>(src, dst, node_x, agg1, deg);
  k_layer1<<<(N_NODES * 32) / 256, 256, 0, stream>>>(agg1, node_x, deg, Wl1, bl1, Wr1, h);
  k_scatter2<<<(N_EDGES * 32) / 256, 256, 0, stream>>>(src, dst, h, agg2);
  k_layer2<<<(BATCH * 64) / 256, 256, 0, stream>>>(x, agg2, h, deg, Wl2, bl2, Wr2, eg);

  // AE branch (WMMA)
  k_encoder<<<BATCH / 16, 128, 0, stream>>>(x, rating, WencH, b_enc, eg, z);
  k_decoder<<<(256 * (NP / 16)) / 8, 256, 0, stream>>>(z, WdecH, b_dec, out);
}
